// MultiHeadAttention_54125177864209
// MI455X (gfx1250) — matrix operand order repaired and checked by reading, not compiled, not run on hardware
//
#include <hip/hip_runtime.h>
#include <hip/hip_bf16.h>
#include <math.h>

// Problem constants (from reference)
#define DM   1024
#define NH   16
#define DK   64
#define BB   2
#define LL   2048

typedef __attribute__((ext_vector_type(16))) __bf16 v16bf;
typedef __attribute__((ext_vector_type(8)))  __bf16 v8bf;
typedef __attribute__((ext_vector_type(8)))  float  v8f;
typedef __attribute__((ext_vector_type(4)))  float  v4f;

__device__ __forceinline__ v8f wmma_bf16(v16bf a, v16bf b, v8f c) {
  // D = A(16x32 bf16) x B(32x16 bf16) + C(16x16 f32)
  return __builtin_amdgcn_wmma_f32_16x16x32_bf16(
      /*neg_a=*/false, a, /*neg_b=*/false, b,
      /*c_mod=*/(short)0, c, /*reuse_a=*/false, /*reuse_b=*/false);
}

// Load 8 consecutive elements into fragment positions [base, base+8).
__device__ __forceinline__ void ld8(const __bf16* p, v16bf& f, int base) {
  v8bf t = *(const v8bf*)p;
#pragma unroll
  for (int i = 0; i < 8; ++i) f[base + i] = t[i];
}
__device__ __forceinline__ void ld8(const float* p, v16bf& f, int base) {
  v4f t0 = *(const v4f*)p;
  v4f t1 = *(const v4f*)(p + 4);
#pragma unroll
  for (int i = 0; i < 4; ++i) {
    f[base + i]     = (__bf16)t0[i];
    f[base + 4 + i] = (__bf16)t1[i];
  }
}

// ---------------------------------------------------------------------------
// Elementwise f32 -> bf16 conversion (8 elements/thread).
// ---------------------------------------------------------------------------
__global__ __launch_bounds__(256) void cvt_kernel(const float* __restrict__ src,
                                                  __bf16* __restrict__ dst, int n8)
{
  const int i = blockIdx.x * blockDim.x + threadIdx.x;
  if (i >= n8) return;
  v4f a = *(const v4f*)(src + (size_t)i * 8);
  v4f b = *(const v4f*)(src + (size_t)i * 8 + 4);
  v8bf o;
#pragma unroll
  for (int j = 0; j < 4; ++j) { o[j] = (__bf16)a[j]; o[4 + j] = (__bf16)b[j]; }
  *(v8bf*)(dst + (size_t)i * 8) = o;
}

// ---------------------------------------------------------------------------
// Kernel 1: QKV projection.  C[m,n] = X[m,:] . W[n,:] + bias[n]
// One wave -> one 64x64 C block (4x4 WMMA tiles, 16 wmma / K-step).
// Q,K written [B,H,L,64]; V written transposed [B,H,64,L].
// ---------------------------------------------------------------------------
template <typename TX, typename TW>
__global__ __launch_bounds__(256) void qkv_proj_kernel(
    const TX* __restrict__ Xq, const TX* __restrict__ Xk, const TX* __restrict__ Xv,
    const TW* __restrict__ Wq, const TW* __restrict__ Wk, const TW* __restrict__ Wv,
    const float* __restrict__ bq, const float* __restrict__ bk,
    const float* __restrict__ bv,
    __bf16* __restrict__ Qo, __bf16* __restrict__ Ko, __bf16* __restrict__ Vt)
{
  const int wave = blockIdx.x * (blockDim.x >> 5) + (threadIdx.x >> 5);
  const int lane = threadIdx.x & 31;
  const int r  = lane & 15;      // A row / B col / D col within 16x16 tile
  const int hh = lane >> 4;      // lane half

  // tasks: 3 * (4096/64) * (1024/64) = 3 * 64 * 16
  const int z   = wave / (64 * 16);
  const int rem = wave % (64 * 16);
  const int m0  = (rem / 16) * 64;
  const int n0  = (rem % 16) * 64;

  const TX* X = (z == 0) ? Xq : (z == 1) ? Xk : Xv;
  const TW* W = (z == 0) ? Wq : (z == 1) ? Wk : Wv;
  const float* bias = (z == 0) ? bq : (z == 1) ? bk : bv;

  const TX* xrow[4];
  const TW* wrow[4];
#pragma unroll
  for (int i = 0; i < 4; ++i) {
    xrow[i] = X + (size_t)(m0 + 16 * i + r) * DM;   // A rows
    wrow[i] = W + (size_t)(n0 + 16 * i + r) * DM;   // B cols (= W rows)
  }

  v8f acc[4][4] = {};
  for (int kk = 0; kk < DM; kk += 32) {
    if (kk + 256 < DM) {
      __builtin_prefetch(xrow[0] + kk + 256, 0, 1);   // global_prefetch_b8
      __builtin_prefetch(wrow[0] + kk + 256, 0, 1);
    }
    v16bf af[4], bf[4];
#pragma unroll
    for (int i = 0; i < 4; ++i) {
      // A frag: elems 0..7 -> k = kk+8h+e ; elems 8..15 -> k = kk+16+8h+e
      ld8(xrow[i] + kk + 8 * hh,      af[i], 0);
      ld8(xrow[i] + kk + 16 + 8 * hh, af[i], 8);
      // B frag: elems 0..7 -> k = kk+8h+e ; elems 8..15 -> k = kk+16+8h+e
      ld8(wrow[i] + kk + 8 * hh,      bf[i], 0);
      ld8(wrow[i] + kk + 16 + 8 * hh, bf[i], 8);
    }
#pragma unroll
    for (int mi = 0; mi < 4; ++mi)
#pragma unroll
      for (int ni = 0; ni < 4; ++ni)
        acc[mi][ni] = wmma_bf16(af[mi], bf[ni], acc[mi][ni]);
  }

#pragma unroll
  for (int ni = 0; ni < 4; ++ni) {
    const int n    = n0 + 16 * ni + r;
    const float bn = bias[n];
    const int head = n >> 6;
    const int dd   = n & 63;
#pragma unroll
    for (int mi = 0; mi < 4; ++mi) {
#pragma unroll
      for (int d = 0; d < 8; ++d) {
        const int m  = m0 + 16 * mi + d + 8 * hh;   // global row = b*L + l
        const int bi = m >> 11;
        const int l  = m & (LL - 1);
        const float v = acc[mi][ni][d] + bn;
        if (z == 0)      Qo[(((size_t)(bi * NH + head) * LL + l) << 6) + dd] = (__bf16)v;
        else if (z == 1) Ko[(((size_t)(bi * NH + head) * LL + l) << 6) + dd] = (__bf16)v;
        else             Vt[((size_t)(bi * NH + head) * DK + dd) * LL + l]   = (__bf16)v;
      }
    }
  }
}

// ---------------------------------------------------------------------------
// Kernel 2: RoPE in-place on Q,K ([B,H,L,64] bf16). One thread per (b,h,l,pair).
// ---------------------------------------------------------------------------
__global__ __launch_bounds__(256) void rope_kernel(__bf16* __restrict__ Q,
                                                   __bf16* __restrict__ K)
{
  const int idx = blockIdx.x * blockDim.x + threadIdx.x;   // 2^21 total
  if (idx >= BB * NH * LL * 32) return;
  const int i  = idx & 31;
  const int l  = (idx >> 5) & (LL - 1);
  const int bh = idx >> 16;
  const size_t base = ((size_t)bh * LL + l) * DK;

  // inv_freq = 10000^(-i/32) = 2^(-i * log2(10000)/32)
  const float inv = exp2f(-(float)i * (13.287712379549449f / 32.0f));
  const float ang = (float)l * inv;
  const float c = cosf(ang), s = sinf(ang);

  float q1 = (float)Q[base + i], q2 = (float)Q[base + i + 32];
  Q[base + i]      = (__bf16)(q1 * c - q2 * s);
  Q[base + i + 32] = (__bf16)(q2 * c + q1 * s);
  float k1 = (float)K[base + i], k2 = (float)K[base + i + 32];
  K[base + i]      = (__bf16)(k1 * c - k2 * s);
  K[base + i + 32] = (__bf16)(k2 * c + k1 * s);
}

// ---------------------------------------------------------------------------
// Kernel 3: causal flash attention.  One wave -> 16 query rows of one (b,h).
// Computes S^T = K * Q^T so the D-fragment layout (col=query) makes softmax
// reductions 1 shuffle, and S^T's D layout == P's A layout for P*V (no LDS).
// All 8 fragment loads of a key block are issued before any consuming wmma so
// the softmax VALU work overlaps the in-flight V loads.
// ---------------------------------------------------------------------------
__global__ __launch_bounds__(256) void attn_kernel(
    const __bf16* __restrict__ Qo, const __bf16* __restrict__ Ko,
    const __bf16* __restrict__ Vo, __bf16* __restrict__ ctx)
{
  const int wave = blockIdx.x * (blockDim.x >> 5) + (threadIdx.x >> 5);
  const int lane = threadIdx.x & 31;
  const int r  = lane & 15;
  const int hh = lane >> 4;

  const int qt = wave & 127;          // 128 query tiles of 16
  const int bh = wave >> 7;           // 32 (b,h) pairs
  const int b  = bh >> 4, h = bh & 15;
  const int qb = qt << 4;

  const __bf16* Qp = Qo + (size_t)bh * LL * DK;
  const __bf16* Kp = Ko + (size_t)bh * LL * DK;
  const __bf16* Vp = Vo + (size_t)bh * DK * LL;   // [64, L]

  // Q^T B-fragments (col = query = r, elem e<8 -> dim 32*s2 + 8*hh + e, e>=8 -> +16).
  v16bf qf[2];
#pragma unroll
  for (int s2 = 0; s2 < 2; ++s2) {
    const __bf16* p = Qp + (size_t)(qb + r) * DK + 32 * s2 + 8 * hh;
    ld8(p,      qf[s2], 0);
    ld8(p + 16, qf[s2], 8);
  }

  v8f   oacc[4] = {{}, {}, {}, {}};
  float m_run = -1e30f, l_run = 0.0f;
  const float scale = 0.125f;                 // 1/sqrt(64)
  const int   qidx  = qb + r;                 // this lane's query column

  for (int kb = 0; kb < qb + 16; kb += 32) {
    // ---- Issue ALL loads for this key block first (4 K frags + 4 V frags).
    v16bf ka[2][2];   // [key tile t][k-step s2]: A = K rows (row = key)
#pragma unroll
    for (int t = 0; t < 2; ++t)
#pragma unroll
      for (int s2 = 0; s2 < 2; ++s2) {
        const __bf16* p = Kp + (size_t)(kb + 16 * t + r) * DK + 32 * s2 + 8 * hh;
        ld8(p,      ka[t][s2], 0);
        ld8(p + 16, ka[t][s2], 8);
      }
    v16bf vb[4];      // B = V^T frag (col = dim-in-group, elem e<8 -> key 8h+e, e>=8 -> 16+8h+e-8)
#pragma unroll
    for (int g = 0; g < 4; ++g) {
      const __bf16* p = Vp + (size_t)(16 * g + r) * LL + kb + 8 * hh;
      ld8(p,      vb[g], 0);
      ld8(p + 16, vb[g], 8);
    }

    // ---- S^T tiles: two K-steps over the 64 feature dims.
    v8f st[2];
#pragma unroll
    for (int t = 0; t < 2; ++t) {
      v8f sacc = {};
      sacc = wmma_bf16(ka[t][0], qf[0], sacc);
      sacc = wmma_bf16(ka[t][1], qf[1], sacc);
      st[t] = sacc;
    }

    // ---- Mask + scale + online softmax. Lane holds keys kb+d+8h (t=0), +16 (t=1).
    float p0[8], p1[8];
    float mloc = -1e30f;
#pragma unroll
    for (int d = 0; d < 8; ++d) {
      const int k0 = kb + d + 8 * hh;
      p0[d] = (k0      <= qidx) ? st[0][d] * scale : -INFINITY;
      p1[d] = (k0 + 16 <= qidx) ? st[1][d] * scale : -INFINITY;
      mloc = fmaxf(mloc, fmaxf(p0[d], p1[d]));
    }
    mloc = fmaxf(mloc, __shfl_xor(mloc, 16, 32));      // combine lane halves
    const float m_new = fmaxf(m_run, mloc);
    const float alpha = __expf(m_run - m_new);

    float lsum = 0.0f;
    v16bf pa;                                          // P as A-fragment
#pragma unroll
    for (int d = 0; d < 8; ++d) {
      const float e0 = __expf(p0[d] - m_new);
      const float e1 = __expf(p1[d] - m_new);
      lsum += e0 + e1;
      pa[d]     = (__bf16)e0;      // key-in-block 8h+d    -> elem d
      pa[8 + d] = (__bf16)e1;      // key-in-block 16+8h+d -> elem 8+d
    }
    lsum += __shfl_xor(lsum, 16, 32);
    l_run = l_run * alpha + lsum;
    m_run = m_new;

    // ---- Rescale O rows (row q = d + 8*hh): fetch alpha for that query.
    float ar[8];
#pragma unroll
    for (int d = 0; d < 8; ++d) ar[d] = __shfl(alpha, d + 8 * hh, 32);
#pragma unroll
    for (int g = 0; g < 4; ++g)
#pragma unroll
      for (int d = 0; d < 8; ++d) oacc[g][d] *= ar[d];

    // ---- O += P * V
#pragma unroll
    for (int g = 0; g < 4; ++g) oacc[g] = wmma_bf16(pa, vb[g], oacc[g]);
  }

  // Epilogue: divide by row sums, write ctx [B, L, H*64] bf16.
  const float rs = (l_run > 0.0f) ? 1.0f / l_run : 0.0f;
  float rr[8];
#pragma unroll
  for (int d = 0; d < 8; ++d) rr[d] = __shfl(rs, d + 8 * hh, 32);
#pragma unroll
  for (int g = 0; g < 4; ++g)
#pragma unroll
    for (int d = 0; d < 8; ++d) {
      const int lq = qb + d + 8 * hh;
      ctx[(size_t)(b * LL + lq) * DM + h * DK + 16 * g + r] =
          (__bf16)(oacc[g][d] * rr[d]);
    }
}

// ---------------------------------------------------------------------------
// Kernel 4: output projection. out[m,n] = ctx[m,:] . Wo[n,:] + bo[n]  (f32 out)
// One wave -> one 64x64 block (16 wmma / K-step).
// ---------------------------------------------------------------------------
template <typename TW>
__global__ __launch_bounds__(256) void out_proj_kernel(
    const __bf16* __restrict__ Co, const TW* __restrict__ Wo,
    const float* __restrict__ bo, float* __restrict__ out)
{
  const int wave = blockIdx.x * (blockDim.x >> 5) + (threadIdx.x >> 5);
  const int lane = threadIdx.x & 31;
  const int r  = lane & 15;
  const int hh = lane >> 4;

  const int m0 = (wave / 16) * 64;
  const int n0 = (wave % 16) * 64;

  const __bf16* arow[4];
  const TW*     wrow[4];
#pragma unroll
  for (int i = 0; i < 4; ++i) {
    arow[i] = Co + (size_t)(m0 + 16 * i + r) * DM;
    wrow[i] = Wo + (size_t)(n0 + 16 * i + r) * DM;
  }

  v8f acc[4][4] = {};
  for (int kk = 0; kk < DM; kk += 32) {
    if (kk + 256 < DM) {
      __builtin_prefetch(arow[0] + kk + 256, 0, 1);
      __builtin_prefetch(wrow[0] + kk + 256, 0, 1);
    }
    v16bf af[4], bf[4];
#pragma unroll
    for (int i = 0; i < 4; ++i) {
      ld8(arow[i] + kk + 8 * hh,      af[i], 0);
      ld8(arow[i] + kk + 16 + 8 * hh, af[i], 8);
      ld8(wrow[i] + kk + 8 * hh,      bf[i], 0);
      ld8(wrow[i] + kk + 16 + 8 * hh, bf[i], 8);
    }
#pragma unroll
    for (int mi = 0; mi < 4; ++mi)
#pragma unroll
      for (int ni = 0; ni < 4; ++ni)
        acc[mi][ni] = wmma_bf16(af[mi], bf[ni], acc[mi][ni]);
  }

#pragma unroll
  for (int ni = 0; ni < 4; ++ni) {
    const int n    = n0 + 16 * ni + r;
    const float bn = bo[n];
#pragma unroll
    for (int mi = 0; mi < 4; ++mi)
#pragma unroll
      for (int d = 0; d < 8; ++d)
        out[(size_t)(m0 + 16 * mi + d + 8 * hh) * DM + n] = acc[mi][ni][d] + bn;
  }
}

// ---------------------------------------------------------------------------
extern "C" void kernel_launch(void* const* d_in, const int* in_sizes, int n_in,
                              void* d_out, int out_size, void* d_ws, size_t ws_size,
                              hipStream_t stream)
{
  const float* q  = (const float*)d_in[0];
  const float* k  = (const float*)d_in[1];
  const float* v  = (const float*)d_in[2];
  // d_in[3] = causal mask (bool) -- computed analytically in attn_kernel
  const float* Wq = (const float*)d_in[4];
  const float* bq = (const float*)d_in[5];
  const float* Wk = (const float*)d_in[6];
  const float* bk = (const float*)d_in[7];
  const float* Wv = (const float*)d_in[8];
  const float* bv = (const float*)d_in[9];
  const float* Wo = (const float*)d_in[10];
  const float* bo = (const float*)d_in[11];
  float* out = (float*)d_out;

  char* ws = (char*)d_ws;
  const size_t MB = (size_t)1 << 20;
  __bf16* Qb  = (__bf16*)(ws);              // 8 MB  [B,H,L,64]
  __bf16* Kb  = (__bf16*)(ws + 8 * MB);     // 8 MB  [B,H,L,64]
  __bf16* Vt  = (__bf16*)(ws + 16 * MB);    // 8 MB  [B,H,64,L]
  __bf16* ctx = (__bf16*)(ws + 24 * MB);    // 8 MB  [B,L,1024] (aliases Xqb)

  const bool fast = ws_size >= 56 * MB;

  if (fast) {
    // bf16 staging: X copies (Xqb aliases ctx region; consumed before attn) + weights
    __bf16* Xqb = (__bf16*)(ws + 24 * MB);
    __bf16* Xkb = (__bf16*)(ws + 32 * MB);
    __bf16* Xvb = (__bf16*)(ws + 40 * MB);
    __bf16* Wqb = (__bf16*)(ws + 48 * MB);
    __bf16* Wkb = (__bf16*)(ws + 50 * MB);
    __bf16* Wvb = (__bf16*)(ws + 52 * MB);
    __bf16* Wob = (__bf16*)(ws + 54 * MB);

    const int nX8 = BB * LL * DM / 8;   // 524288
    const int nW8 = DM * DM / 8;        // 131072
    cvt_kernel<<<nX8 / 256, 256, 0, stream>>>(q,  Xqb, nX8);
    cvt_kernel<<<nX8 / 256, 256, 0, stream>>>(k,  Xkb, nX8);
    cvt_kernel<<<nX8 / 256, 256, 0, stream>>>(v,  Xvb, nX8);
    cvt_kernel<<<nW8 / 256, 256, 0, stream>>>(Wq, Wqb, nW8);
    cvt_kernel<<<nW8 / 256, 256, 0, stream>>>(Wk, Wkb, nW8);
    cvt_kernel<<<nW8 / 256, 256, 0, stream>>>(Wv, Wvb, nW8);
    cvt_kernel<<<nW8 / 256, 256, 0, stream>>>(Wo, Wob, nW8);

    qkv_proj_kernel<__bf16, __bf16><<<(3 * 64 * 16) / 8, 256, 0, stream>>>(
        Xqb, Xkb, Xvb, Wqb, Wkb, Wvb, bq, bk, bv, Qb, Kb, Vt);
  } else {
    qkv_proj_kernel<float, float><<<(3 * 64 * 16) / 8, 256, 0, stream>>>(
        q, k, v, Wq, Wk, Wv, bq, bk, bv, Qb, Kb, Vt);
  }

  rope_kernel<<<(BB * NH * LL * 32) / 256, 256, 0, stream>>>(Qb, Kb);
  attn_kernel<<<4096 / 8, 256, 0, stream>>>(Qb, Kb, Vt, ctx);

  if (fast) {
    const __bf16* Wob = (const __bf16*)(ws + 54 * MB);
    out_proj_kernel<__bf16><<<(64 * 16) / 8, 256, 0, stream>>>(ctx, Wob, bo, out);
  } else {
    out_proj_kernel<float><<<(64 * 16) / 8, 256, 0, stream>>>(ctx, Wo, bo, out);
  }
}
